// ScalarMLPGAT_28089086116667
// MI455X (gfx1250) — matrix operand order repaired and checked by reading, not compiled, not run on hardware
//
#include <hip/hip_runtime.h>
#include <hip/hip_bf16.h>

// ---------------------------------------------------------------------------
// CDNA5 / gfx1250 implementation.
//   K0  convert fp32 -> bf16 (weights get MLP scales folded in)
//   K1  h = edge_features @ W        (bf16 WMMA, fp32 accum)   [8192x1024x1024]
//   K2  al_s/al_d = h . att_src/dst  (per-row reduction)
//   K3  deterministic per-source-node buckets (no atomics)
//   K4  sparse masked softmax + weighted gather of h rows (+bias)
//   K5  mlp: silu(x@w0s) @ w1s       (two bf16 WMMA GEMMs, silu fused)
//
// GEMM: 256 thr / 8 waves, block tile 128x256, BK=32; each wave owns a 64x64
// sub-tile = 4x4 v_wmma_f32_16x16x32_bf16. LDS layouts are chosen so every
// fragment is two aligned ds_load_b128 (no register shuffling):
//   A: [m][k] halves, rows padded to 20 dwords
//   B: k-pair-packed, n-major: dword [n*20 + k/2] = (k,n) | (k+1,n)<<16
// Register double-buffering + global_prefetch hide global latency.
// ---------------------------------------------------------------------------

typedef __attribute__((ext_vector_type(16))) __bf16 v16bf;
typedef __attribute__((ext_vector_type(8)))  float  v8f;

__device__ __forceinline__ unsigned short f2bf(float f) {
  unsigned int u = __builtin_bit_cast(unsigned int, f);
  unsigned int r = (u + 0x7FFFu + ((u >> 16) & 1u)) >> 16;  // RNE
  return (unsigned short)r;
}

// ---------------------------- K0: fp32 -> bf16 -----------------------------
__global__ __launch_bounds__(256) void convert_bf16_kernel(
    const float* __restrict__ in, unsigned short* __restrict__ out,
    int n, float scale) {
  int i = blockIdx.x * 256 + threadIdx.x;
  if (i < n) out[i] = f2bf(in[i] * scale);
}

// ------------------------ K1/K5: bf16 WMMA GEMM ----------------------------
#define BM 128
#define BNT 256
#define BK 32
#define ADW 20  // dwords per A row (16 data + 4 pad): b128-aligned, bank-spread
#define BDW 20  // dwords per packed-B n-row (16 kp + 4 pad)

__global__ __launch_bounds__(256) void gemm_bf16_kernel(
    const unsigned short* __restrict__ A, const unsigned short* __restrict__ B,
    float* __restrict__ Cf, unsigned short* __restrict__ Cb,
    int M, int N, int K) {
  __shared__ unsigned int sA[BM * ADW];   // [m][k/2] half-pairs, row-major
  __shared__ unsigned int sB[BNT * BDW];  // [n][k/2] packed pairs, n-major

  const int tid  = threadIdx.x;
  const int lane = tid & 31;
  const int wid  = tid >> 5;
  const int m0 = blockIdx.y * BM;
  const int n0 = blockIdx.x * BNT;
  const int wm = (wid & 1) * 64;   // wave M offset in block tile
  const int wn = (wid >> 1) * 64;  // wave N offset in block tile

  const int r15   = lane & 15;
  const int khalf = lane >> 4;

  v8f acc[4][4];
#pragma unroll
  for (int mt = 0; mt < 4; ++mt)
#pragma unroll
    for (int nt = 0; nt < 4; ++nt)
#pragma unroll
      for (int g = 0; g < 8; ++g) acc[mt][nt][g] = 0.0f;

  // loader mappings
  const int arow = tid >> 1, aseg = tid & 1;   // A: 128 rows x 2 segs of 16
  const int bkp  = tid >> 4, bseg = tid & 15;  // B: 16 kp-rows x 16 segs of 16n

  union Stage { uint4 q[2]; unsigned int w[8]; };
  Stage pa, pbl, pbh;  // register double buffer

  const size_t aoff = (size_t)(m0 + arow) * K + aseg * 16;  // + k0
  auto load_tile = [&](int k0) {
    const uint4* ap = reinterpret_cast<const uint4*>(A + aoff + k0);
    pa.q[0] = ap[0];
    pa.q[1] = ap[1];
    const uint4* bp0 = reinterpret_cast<const uint4*>(
        B + (size_t)(k0 + 2 * bkp) * N + n0 + bseg * 16);
    const uint4* bp1 = reinterpret_cast<const uint4*>(
        B + (size_t)(k0 + 2 * bkp + 1) * N + n0 + bseg * 16);
    pbl.q[0] = bp0[0];
    pbl.q[1] = bp0[1];
    pbh.q[0] = bp1[0];
    pbh.q[1] = bp1[1];
  };

  union Frag { uint4 q[2]; unsigned int u[8]; v16bf v; };

  load_tile(0);  // prologue prefetch

  for (int k0 = 0; k0 < K; k0 += BK) {
    __syncthreads();  // previous iteration's fragment reads are done
    // commit prefetched tile to LDS
    {
      uint4* sp = reinterpret_cast<uint4*>(sA) + arow * (ADW / 4) + aseg * 2;
      sp[0] = pa.q[0];
      sp[1] = pa.q[1];
#pragma unroll
      for (int j = 0; j < 8; ++j) {
        // repack: [n][kp] dword = lo(row k, n) | lo(row k+1, n) << 16
        sB[(bseg * 16 + 2 * j + 0) * BDW + bkp] =
            (pbl.w[j] & 0xFFFFu) | (pbh.w[j] << 16);
        sB[(bseg * 16 + 2 * j + 1) * BDW + bkp] =
            (pbl.w[j] >> 16) | (pbh.w[j] & 0xFFFF0000u);
      }
    }
    __syncthreads();

    // prefetch next tile while this one computes (register double buffer)
    if (k0 + BK < K) {
      load_tile(k0 + BK);
      if (k0 + 2 * BK < K) {  // L2/L0 warm-up two tiles ahead
        __builtin_prefetch(A + aoff + k0 + 2 * BK, 0, 1);
        __builtin_prefetch(
            B + (size_t)(k0 + 2 * BK + 2 * bkp) * N + n0 + bseg * 16, 0, 1);
      }
    }

    // A fragments: two aligned b128 per tile row group
    Frag af[4];
#pragma unroll
    for (int mt = 0; mt < 4; ++mt) {
      const uint4* sp = reinterpret_cast<const uint4*>(sA) +
                        (wm + mt * 16 + r15) * (ADW / 4) + khalf;
      af[mt].q[0] = sp[0];
      af[mt].q[1] = sp[2];
    }
    // B fragments: two aligned b128, n-major packed pairs
#pragma unroll
    for (int nt = 0; nt < 4; ++nt) {
      Frag bfr;
      const uint4* sp = reinterpret_cast<const uint4*>(sB) +
                        (wn + nt * 16 + r15) * (BDW / 4) + khalf;
      bfr.q[0] = sp[0];
      bfr.q[1] = sp[2];
#pragma unroll
      for (int mt = 0; mt < 4; ++mt)
        acc[mt][nt] = __builtin_amdgcn_wmma_f32_16x16x32_bf16(
            false, af[mt].v, false, bfr.v, (short)0, acc[mt][nt], false, false);
    }
  }

  // Epilogue: C/D layout — VGPR g holds M = g + 8*(lane/16), N = lane%16
#pragma unroll
  for (int mt = 0; mt < 4; ++mt)
#pragma unroll
    for (int nt = 0; nt < 4; ++nt) {
      const int mB = m0 + wm + mt * 16 + khalf * 8;
      const int nB = n0 + wn + nt * 16 + r15;
#pragma unroll
      for (int g = 0; g < 8; ++g) {
        float x = acc[mt][nt][g];
        size_t off = (size_t)(mB + g) * N + nB;
        if (Cb) {  // fused silu -> bf16 (MLP hidden layer)
          float s = x / (1.0f + __expf(-x));
          Cb[off] = f2bf(s);
        } else {
          Cf[off] = x;
        }
      }
    }
}

// ------------------- K2: al_s/al_d row reductions of h ---------------------
__global__ __launch_bounds__(256) void alpha_kernel(
    const float* __restrict__ h, const float* __restrict__ as,
    const float* __restrict__ ad, float* __restrict__ al_s,
    float* __restrict__ al_d, int C) {
  const int e = blockIdx.x;
  __shared__ float ss[256], sd[256];
  float ps = 0.f, pd = 0.f;
  for (int c = threadIdx.x; c < C; c += 256) {
    float v = h[(size_t)e * C + c];
    ps += v * as[c];
    pd += v * ad[c];
  }
  ss[threadIdx.x] = ps;
  sd[threadIdx.x] = pd;
  __syncthreads();
  for (int o = 128; o > 0; o >>= 1) {
    if (threadIdx.x < o) {
      ss[threadIdx.x] += ss[threadIdx.x + o];
      sd[threadIdx.x] += sd[threadIdx.x + o];
    }
    __syncthreads();
  }
  if (threadIdx.x == 0) {
    al_s[e] = ss[0];
    al_d[e] = sd[0];
  }
}

// ----------- K3: deterministic per-source buckets (no atomics) -------------
__global__ __launch_bounds__(1024) void bucket_kernel(
    const int* __restrict__ src, int E, int Nn, int* __restrict__ startG,
    int* __restrict__ cntG, int* __restrict__ bucket) {
  __shared__ int sc[1024];
  __shared__ int so[1024];
  const int s = threadIdx.x;
  int c = 0;
  if (s < Nn)
    for (int e = 0; e < E; ++e) c += (src[e] == s);
  sc[s] = c;
  __syncthreads();
  if (threadIdx.x == 0) {
    int accum = 0;
    for (int i = 0; i < Nn; ++i) { so[i] = accum; accum += sc[i]; }
  }
  __syncthreads();
  if (s < Nn) {
    int p = so[s];
    startG[s] = p;
    cntG[s] = sc[s];
    for (int e = 0; e < E; ++e)
      if (src[e] == s) bucket[p++] = e;
  }
}

// --------- K4: masked softmax over bucket + weighted gather of h -----------
__global__ __launch_bounds__(256) void gat_aggregate_kernel(
    const float* __restrict__ h, const int* __restrict__ src,
    const int* __restrict__ dst, const float* __restrict__ al_s,
    const float* __restrict__ al_d, const int* __restrict__ start,
    const int* __restrict__ cnt, const int* __restrict__ bucket,
    const float* __restrict__ bias, float* __restrict__ out, int C) {
  const int i = blockIdx.x;
  const int tid = threadIdx.x;
  const int s = src[i];
  const bool validi = (s != dst[i]);
  const float adi = al_d[i];
  const int b0 = start[s];
  const int k = cnt[s];

  auto leaky = [](float x) { return x > 0.f ? x : 0.2f * x; };
  const float lself = leaky(al_s[i] + adi);

  __shared__ float red[256];
  float lmax = lself;
  if (validi)
    for (int t = tid; t < k; t += 256) {
      int j = bucket[b0 + t];
      if (j < i && dst[j] != s) lmax = fmaxf(lmax, leaky(al_s[j] + adi));
    }
  red[tid] = lmax;
  __syncthreads();
  for (int o = 128; o > 0; o >>= 1) {
    if (tid < o) red[tid] = fmaxf(red[tid], red[tid + o]);
    __syncthreads();
  }
  lmax = red[0];
  __syncthreads();
  float lsum = 0.f;
  if (validi)
    for (int t = tid; t < k; t += 256) {
      int j = bucket[b0 + t];
      if (j < i && dst[j] != s) lsum += __expf(leaky(al_s[j] + adi) - lmax);
    }
  red[tid] = lsum;
  __syncthreads();
  for (int o = 128; o > 0; o >>= 1) {
    if (tid < o) red[tid] += red[tid + o];
    __syncthreads();
  }
  lsum = red[0] + __expf(lself - lmax);
  const float inv = 1.0f / lsum;

  const int c = tid * 4;  // C == 1024, block == 256
  const size_t rowq = (size_t)C >> 2;
  const float4* h4 = reinterpret_cast<const float4*>(h);
  float4 a4;
  {
    float w = __expf(lself - lmax) * inv;
    float4 hv = h4[(size_t)i * rowq + (c >> 2)];
    a4.x = w * hv.x; a4.y = w * hv.y; a4.z = w * hv.z; a4.w = w * hv.w;
  }
  if (validi)
    for (int t = 0; t < k; ++t) {
      int j = bucket[b0 + t];
      if (j < i && dst[j] != s) {
        float w = __expf(leaky(al_s[j] + adi) - lmax) * inv;
        float4 hv = h4[(size_t)j * rowq + (c >> 2)];
        a4.x += w * hv.x; a4.y += w * hv.y; a4.z += w * hv.z; a4.w += w * hv.w;
      }
    }
  a4.x += bias[c]; a4.y += bias[c + 1]; a4.z += bias[c + 2]; a4.w += bias[c + 3];
  reinterpret_cast<float4*>(out)[(size_t)i * rowq + (c >> 2)] = a4;
}

// ---------------------------------------------------------------------------
extern "C" void kernel_launch(void* const* d_in, const int* in_sizes, int n_in,
                              void* d_out, int out_size, void* d_ws,
                              size_t ws_size, hipStream_t stream) {
  const float* edge_f = (const float*)d_in[0];
  const float* node_f = (const float*)d_in[1];
  const int*   eidx   = (const int*)d_in[2];
  const float* Wg     = (const float*)d_in[3];
  const float* a_src  = (const float*)d_in[4];
  const float* a_dst  = (const float*)d_in[5];
  const float* bias   = (const float*)d_in[6];
  const float* w0     = (const float*)d_in[7];
  const float* w1     = (const float*)d_in[8];

  const int C  = in_sizes[4];          // 1024
  const int E  = in_sizes[0] / C;      // 8192
  const int Nn = in_sizes[1] / C;      // 512
  const int* src = eidx;
  const int* dst = eidx + E;

  float* out_edge = (float*)d_out;
  float* out_node = (float*)d_out + (size_t)E * C;

  char* wp = (char*)d_ws;
  auto carve = [&](size_t bytes) -> void* {
    void* p = (void*)wp;
    wp += (bytes + 255) & ~(size_t)255;
    return p;
  };
  float*          h   = (float*)carve((size_t)E * C * sizeof(float));
  unsigned short* Ab  = (unsigned short*)carve((size_t)E * C * 2);   // edge bf16
  unsigned short* Wgb = (unsigned short*)carve((size_t)C * C * 2);
  unsigned short* W0b = (unsigned short*)carve((size_t)C * C * 2);
  unsigned short* W1b = (unsigned short*)carve((size_t)C * C * 2);
  unsigned short* Xb  = (unsigned short*)carve((size_t)Nn * C * 2);  // node bf16
  unsigned short* T1b = (unsigned short*)carve((size_t)Nn * C * 2);  // silu mid
  float* al_s = (float*)carve((size_t)E * sizeof(float));
  float* al_d = (float*)carve((size_t)E * sizeof(float));
  int*   bstart = (int*)carve((size_t)Nn * sizeof(int));
  int*   bcnt   = (int*)carve((size_t)Nn * sizeof(int));
  int*   bucket = (int*)carve((size_t)E * sizeof(int));

  const float s0 = 1.0f / sqrtf((float)C);        // mlp_w0 scale
  const float s1 = 1.6790524f / sqrtf((float)C);  // SILU_CST / sqrt(C)

  auto cdiv = [](int a, int b) { return (a + b - 1) / b; };

  // K0: conversions
  convert_bf16_kernel<<<cdiv(E * C, 256), 256, 0, stream>>>(edge_f, Ab, E * C, 1.0f);
  convert_bf16_kernel<<<cdiv(C * C, 256), 256, 0, stream>>>(Wg, Wgb, C * C, 1.0f);
  convert_bf16_kernel<<<cdiv(C * C, 256), 256, 0, stream>>>(w0, W0b, C * C, s0);
  convert_bf16_kernel<<<cdiv(C * C, 256), 256, 0, stream>>>(w1, W1b, C * C, s1);
  convert_bf16_kernel<<<cdiv(Nn * C, 256), 256, 0, stream>>>(node_f, Xb, Nn * C, 1.0f);

  // K1: h = edge_features @ W   (WMMA)
  gemm_bf16_kernel<<<dim3(C / BNT, E / BM), 256, 0, stream>>>(
      Ab, Wgb, h, nullptr, E, C, C);

  // K2: attention logit pieces
  alpha_kernel<<<E, 256, 0, stream>>>(h, a_src, a_dst, al_s, al_d, C);

  // K3: deterministic buckets
  bucket_kernel<<<1, Nn, 0, stream>>>(src, E, Nn, bstart, bcnt, bucket);

  // K4: sparse masked softmax + gather -> new_edge_features
  gat_aggregate_kernel<<<E, 256, 0, stream>>>(h, src, dst, al_s, al_d, bstart,
                                              bcnt, bucket, bias, out_edge, C);

  // K5: MLP  silu(x@w0s) @ w1s  (WMMA x2, silu fused into first epilogue)
  gemm_bf16_kernel<<<dim3(C / BNT, Nn / BM), 256, 0, stream>>>(
      Xb, W0b, nullptr, T1b, Nn, C, C);
  gemm_bf16_kernel<<<dim3(C / BNT, Nn / BM), 256, 0, stream>>>(
      T1b, W1b, out_node, nullptr, Nn, C, C);
}
